// Mid_Transformer_Golobal_layer_60344290509340
// MI455X (gfx1250) — matrix operand order repaired and checked by reading, not compiled, not run on hardware
//
#include <hip/hip_runtime.h>
#include <hip/hip_bf16.h>
#include <math.h>

typedef __attribute__((ext_vector_type(16))) __bf16 v16bf;
typedef __attribute__((ext_vector_type(8)))  __bf16 v8bf;
typedef __attribute__((ext_vector_type(8)))  float  v8f;

// ---------- bf16 <-> f32 helpers ----------
static __device__ inline float b2f(__bf16 x){
  unsigned short u = __builtin_bit_cast(unsigned short, x);
  unsigned int v = ((unsigned int)u) << 16;
  return __builtin_bit_cast(float, v);
}
static __device__ inline __bf16 f2b(float f){
  unsigned int u = __builtin_bit_cast(unsigned int, f);
  unsigned int r = u + 0x7FFFu + ((u >> 16) & 1u);   // RNE
  unsigned short h = (unsigned short)(r >> 16);
  return __builtin_bit_cast(__bf16, h);
}

// ---------- vectorized WMMA fragment constructors (all LDS reads are b128) ----------
static __device__ inline v16bf frag16(const __bf16* p){        // 16 contiguous bf16
  v8bf a = *(const v8bf*)p, b = *(const v8bf*)(p + 8);
  v16bf r;
  #pragma unroll
  for (int e=0;e<8;++e){ r[e]=a[e]; r[e+8]=b[e]; }
  return r;
}
static __device__ inline v16bf fragA16(const __bf16* p){       // p..p+7, p+16..p+23
  v8bf a = *(const v8bf*)p, b = *(const v8bf*)(p + 16);
  v16bf r;
  #pragma unroll
  for (int e=0;e<8;++e){ r[e]=a[e]; r[e+8]=b[e]; }
  return r;
}
// A: 16x32 bf16 tile, row-major with stride ld (lane-half contiguous K: k = 16h + s)
static __device__ inline v16bf ldsA(const __bf16* base, int ld, int lane){
  int m = lane & 15; int kh = (lane >> 4) << 4;
  return frag16(base + m*ld + kh);
}
// B fragment from an N-major tile (tile[n][k], stride ld): contiguous per-lane reads
static __device__ inline v16bf ldsBrow(const __bf16* base, int ld, int lane){
  int n = lane & 15; int k0 = (lane >> 4) << 4;
  return frag16(base + n*ld + k0);
}
static __device__ inline v8f wmma_bf16(v16bf a, v16bf b, v8f c){
  return __builtin_amdgcn_wmma_f32_16x16x32_bf16(false, a, false, b, (short)0, c, false, false);
}

// ---------- small utility kernels ----------
__global__ void k_add(float* __restrict__ y, const float* __restrict__ xs, int n){
  int i = blockIdx.x*256 + threadIdx.x;
  if (i < n) y[i] += xs[i];
}
// f32 (K,N) weight -> bf16 TRANSPOSED (N,K) for N-major GEMM B tiles
__global__ void k_f2bT(const float* __restrict__ src, __bf16* __restrict__ dst, int K, int N){
  __shared__ float tile[32][33];
  int n0 = blockIdx.x << 5, k0 = blockIdx.y << 5;
  int tx = threadIdx.x, ty = threadIdx.y;
  #pragma unroll
  for (int r=0;r<4;++r)
    tile[ty + 8*r][tx] = src[(size_t)(k0 + ty + 8*r)*N + n0 + tx];
  __syncthreads();
  #pragma unroll
  for (int r=0;r<4;++r)
    dst[(size_t)(n0 + ty + 8*r)*K + k0 + tx] = f2b(tile[tx][ty + 8*r]);
}
// (B,C,H,W) f32 -> (B, T, C) f32
__global__ void k_chw2tc(const float* __restrict__ x, float* __restrict__ y){
  __shared__ float tile[32][33];
  int hw0 = blockIdx.x << 5, c0 = blockIdx.y << 5, b = blockIdx.z;
  int tx = threadIdx.x, ty = threadIdx.y;
  const float* xb = x + (size_t)b*1048576;
  float* yb = y + (size_t)b*1048576;
  #pragma unroll
  for (int r=0;r<4;++r)
    tile[ty + 8*r][tx] = xb[(size_t)(c0 + ty + 8*r)*4096 + hw0 + tx];
  __syncthreads();
  #pragma unroll
  for (int r=0;r<4;++r)
    yb[(size_t)(hw0 + ty + 8*r)*256 + c0 + tx] = tile[tx][ty + 8*r];
}
// (B, T, C) f32 -> (B,C,H,W) f32
__global__ void k_tc2chw(const float* __restrict__ y, float* __restrict__ out){
  __shared__ float tile[32][33];
  int hw0 = blockIdx.x << 5, c0 = blockIdx.y << 5, b = blockIdx.z;
  int tx = threadIdx.x, ty = threadIdx.y;
  const float* yb = y + (size_t)b*1048576;
  float* ob = out + (size_t)b*1048576;
  #pragma unroll
  for (int r=0;r<4;++r)
    tile[ty + 8*r][tx] = yb[(size_t)(hw0 + ty + 8*r)*256 + c0 + tx];
  __syncthreads();
  #pragma unroll
  for (int r=0;r<4;++r)
    ob[(size_t)(c0 + ty + 8*r)*4096 + hw0 + tx] = tile[tx][ty + 8*r];
}

// ---------- LayerNorm: one wave per token, bf16 output ----------
__global__ __launch_bounds__(256)
void k_ln(const float* __restrict__ src, const float* __restrict__ g,
          const float* __restrict__ be, __bf16* __restrict__ dst,
          float eps, int colmode){
  int wid = threadIdx.x >> 5, lane = threadIdx.x & 31;
  int tok = blockIdx.x * 8 + wid;
  size_t base;
  if (colmode){
    int b = tok >> 12, tt = tok & 4095, j = tt >> 6, i = tt & 63;
    base = (size_t)b*1048576 + (size_t)i*16384 + (size_t)j*256;
  } else base = (size_t)tok * 256;
  float x[8]; float s = 0.f;
  #pragma unroll
  for (int u=0; u<8; ++u){ x[u] = src[base + lane*8 + u]; s += x[u]; }
  #pragma unroll
  for (int o=16;o>0;o>>=1) s += __shfl_xor(s, o, 32);
  float mean = s * (1.0f/256.0f);
  float v = 0.f;
  #pragma unroll
  for (int u=0; u<8; ++u){ float d = x[u]-mean; v += d*d; }
  #pragma unroll
  for (int o=16;o>0;o>>=1) v += __shfl_xor(v, o, 32);
  float rs = rsqrtf(v*(1.0f/256.0f) + eps);
  #pragma unroll
  for (int u=0; u<8; ++u){
    int c = lane*8+u;
    dst[(size_t)tok*256 + c] = f2b((x[u]-mean)*rs*g[c] + be[c]);
  }
}

// ---------- WMMA GEMM, double-buffered LDS, N-major B ----------
// C[MxN] = A[MxK](bf16, row-major) * Bt[NxK](bf16, N-major) + bias
#define GLDA 40
#define GLDB 40
#define ASZ (128*GLDA)
#define BSZ (64*GLDB)
__global__ __launch_bounds__(256)
void k_gemm(const __bf16* __restrict__ A, const __bf16* __restrict__ Bt,
            const float* __restrict__ bias, float* __restrict__ outf,
            __bf16* __restrict__ outb, int M, int N, int K, int mode){
  __shared__ __align__(16) __bf16 As[2*ASZ];
  __shared__ __align__(16) __bf16 Bs[2*BSZ];
  const int t = threadIdx.x;
  const int lane = t & 31;
  const int wid = t >> 5;
  const int mw = wid >> 1, nw = wid & 1;
  const int m0 = blockIdx.x * 128;
  const int n0 = blockIdx.y * 64;
  v8f acc[2][2] = {};

  const int ar = t >> 1, ac = (t & 1) << 4;      // A: 128 rows x 32, two v8bf per thread
  const int br = t >> 2, bc = (t & 3) << 3;      // Bt: 64 rows x 32, one v8bf per thread

  auto loadTile = [&](int k0, int buf){
    __bf16* as = As + buf*ASZ;
    __bf16* bs = Bs + buf*BSZ;
    *(v8bf*)(&as[ar*GLDA + ac    ]) = *(const v8bf*)(A  + (size_t)(m0 + ar)*K + k0 + ac);
    *(v8bf*)(&as[ar*GLDA + ac + 8]) = *(const v8bf*)(A  + (size_t)(m0 + ar)*K + k0 + ac + 8);
    *(v8bf*)(&bs[br*GLDB + bc    ]) = *(const v8bf*)(Bt + (size_t)(n0 + br)*K + k0 + bc);
  };

  int buf = 0;
  loadTile(0, 0);
  for (int k0 = 0; k0 < K; k0 += 32){
    __syncthreads();
    if (k0 + 32 < K) loadTile(k0 + 32, buf ^ 1);   // overlap next tile with WMMA
    const __bf16* as = As + buf*ASZ;
    const __bf16* bs = Bs + buf*BSZ;
    v16bf af[2], bfr[2];
    af[0]  = ldsA(&as[(mw*32     )*GLDA], GLDA, lane);
    af[1]  = ldsA(&as[(mw*32 + 16)*GLDA], GLDA, lane);
    bfr[0] = ldsBrow(&bs[(nw*32     )*GLDB], GLDB, lane);
    bfr[1] = ldsBrow(&bs[(nw*32 + 16)*GLDB], GLDB, lane);
    #pragma unroll
    for (int i=0;i<2;++i)
      #pragma unroll
      for (int j=0;j<2;++j)
        acc[i][j] = wmma_bf16(af[i], bfr[j], acc[i][j]);
    buf ^= 1;
  }
  const int nlo = lane & 15, mhi = (lane >> 4) << 3;
  #pragma unroll
  for (int i=0;i<2;++i){
    #pragma unroll
    for (int j=0;j<2;++j){
      int nn = n0 + nw*32 + j*16 + nlo;
      float bv = bias ? bias[nn] : 0.f;
      #pragma unroll
      for (int v=0; v<8; ++v){
        int mm = m0 + mw*32 + i*16 + mhi + v;
        float val = acc[i][j][v] + bv;
        size_t off = (size_t)mm * N + nn;
        if (mode == 0)      outf[off] = val;
        else if (mode == 1) outf[off] += val;
        else if (mode == 2) outb[off] = f2b(val);
        else if (mode == 3) { float s = 1.f/(1.f+__expf(-1.702f*val));
                              outb[off] = f2b(val*s); }
        else                 outb[off] = f2b(0.5f*val*(1.f+erff(val*0.70710678118f)));
      }
    }
  }
}

// ---------- row/col attention core: one block per (sequence, head), T=64, hs=32 ----------
__global__ __launch_bounds__(128)
void k_rcattn(const __bf16* __restrict__ q, const __bf16* __restrict__ k,
              const __bf16* __restrict__ v, const float* __restrict__ relw,
              __bf16* __restrict__ ao, int colmode){
  __shared__ __align__(16) __bf16 Qs[64*40];
  __shared__ __align__(16) __bf16 Ks[64*40];
  __shared__ __align__(16) __bf16 Vt[32*72];   // V transposed: [hs][token] -> contiguous B frags
  __shared__ float Ss[64*68];
  __shared__ __align__(16) __bf16 Ps[64*72];
  const int head = blockIdx.x & 7;
  const int seq  = blockIdx.x >> 3;
  const int b = seq >> 6, sw = seq & 63;
  const int t = threadIdx.x, lane = t & 31, wv = t >> 5;
  const float* rw = relw + head * 127;
  #pragma unroll
  for (int c = 0; c < 2; ++c){
    int chunk = t*2 + c;
    int p = chunk >> 2, off = (chunk & 3) << 3;
    size_t tok = colmode ? ((size_t)b*4096 + sw*64 + p) : ((size_t)b*4096 + p*64 + sw);
    size_t src = tok*256 + head*32 + off;
    *(v8bf*)(&Qs[p*40+off]) = *(const v8bf*)(q + src);
    *(v8bf*)(&Ks[p*40+off]) = *(const v8bf*)(k + src);
    v8bf vv = *(const v8bf*)(v + src);
    #pragma unroll
    for (int i=0;i<8;++i) Vt[(off+i)*72 + p] = vv[i];   // transpose at fill time
  }
  __syncthreads();
  const int m0 = wv * 16;
  const float scale = 0.17677669529663687f;    // 1/sqrt(32)
  {
    v16bf aq = ldsA(&Qs[m0*40], 40, lane);
    #pragma unroll
    for (int nb = 0; nb < 4; ++nb){
      v16bf bk = ldsBrow(&Ks[nb*16*40], 40, lane);   // Kt tile via row-contiguous reads
      v8f s = {};
      s = wmma_bf16(aq, bk, s);
      int nn = nb*16 + (lane & 15);
      int mr0 = m0 + ((lane >> 4) << 3);
      #pragma unroll
      for (int vv2=0; vv2<8; ++vv2){
        int mr = mr0 + vv2;
        Ss[mr*68 + nn] = s[vv2]*scale + rw[nn - mr + 63];
      }
    }
  }
  __syncthreads();
  if (t < 64){                                  // softmax row t
    float mx = -1e30f;
    for (int j=0;j<64;++j) mx = fmaxf(mx, Ss[t*68+j]);
    float sum = 0.f;
    for (int j=0;j<64;++j) sum += __expf(Ss[t*68+j]-mx);
    float inv = 1.f/sum;
    for (int j=0;j<64;++j) Ps[t*72+j] = f2b(__expf(Ss[t*68+j]-mx)*inv);
  }
  __syncthreads();
  v8f acc[2] = {};
  #pragma unroll
  for (int ks = 0; ks < 64; ks += 32){
    v16bf ap = ldsA(&Ps[m0*72 + ks], 72, lane);
    #pragma unroll
    for (int nb = 0; nb < 2; ++nb){
      // B[k][n] = V[token k][hs n] = Vt[n][k], contiguous in k
      v16bf bv = frag16(&Vt[(nb*16 + (lane & 15))*72 + ks + ((lane >> 4) << 4)]);
      acc[nb] = wmma_bf16(ap, bv, acc[nb]);
    }
  }
  const int nlo = lane & 15, mh = (lane >> 4) << 3;
  #pragma unroll
  for (int nb=0; nb<2; ++nb){
    int nn = nb*16 + nlo;
    #pragma unroll
    for (int vv2=0; vv2<8; ++vv2){
      int p = m0 + mh + vv2;
      size_t tok = colmode ? ((size_t)b*4096 + sw*64 + p) : ((size_t)b*4096 + p*64 + sw);
      ao[tok*256 + head*32 + nn] = f2b(acc[nb][vv2]);
    }
  }
}

// ---------- linear self-attention ----------
__global__ __launch_bounds__(256)
void k_sattn_score(const __bf16* __restrict__ q, const __bf16* __restrict__ k,
                   __bf16* __restrict__ P){
  __shared__ float Ss[32*33];
  int b = blockIdx.x >> 3, h = blockIdx.x & 7;
  int t = threadIdx.x;
  int d = t >> 3;
  int e0 = (t & 7) << 2;
  float acc[4] = {0.f,0.f,0.f,0.f};
  size_t base = ((size_t)b*4096)*256 + h*32;
  for (int tok = 0; tok < 4096; ++tok){
    size_t o = base + (size_t)tok*256;
    float qd = b2f(q[o + d]);
    #pragma unroll
    for (int u=0;u<4;++u) acc[u] += qd * b2f(k[o + e0 + u]);
  }
  const float scale = 0.17677669529663687f;
  #pragma unroll
  for (int u=0;u<4;++u) Ss[d*33 + e0 + u] = acc[u]*scale;
  __syncthreads();
  if (t < 32){
    float mx = -1e30f;
    for (int j=0;j<32;++j) mx = fmaxf(mx, Ss[t*33+j]);
    float sum = 0.f;
    for (int j=0;j<32;++j) sum += __expf(Ss[t*33+j]-mx);
    float inv = 1.f/sum;
    for (int j=0;j<32;++j)
      P[((size_t)blockIdx.x)*1024 + t*32 + j] = f2b(__expf(Ss[t*33+j]-mx)*inv);
  }
}
__global__ __launch_bounds__(256)
void k_sattn_apply(const __bf16* __restrict__ v, const __bf16* __restrict__ P,
                   __bf16* __restrict__ ao){
  size_t gid = (size_t)blockIdx.x*256 + threadIdx.x;
  int c = (int)(gid & 255);
  size_t tok = gid >> 8;
  int h = c >> 5, e = c & 31;
  int b = (int)(tok >> 12);
  const __bf16* vr = v + tok*256 + h*32;
  const __bf16* pr = P + ((size_t)(b*8+h))*1024 + e;
  float acc = 0.f;
  #pragma unroll
  for (int d=0; d<32; ++d) acc += b2f(vr[d]) * b2f(pr[d*32]);
  ao[gid] = f2b(acc);
}

// ---------- host orchestration ----------
extern "C" void kernel_launch(void* const* d_in, const int* in_sizes, int n_in,
                              void* d_out, int out_size, void* d_ws, size_t ws_size,
                              hipStream_t stream){
  (void)in_sizes; (void)n_in; (void)out_size; (void)ws_size;
  const float* x    = (const float*)d_in[0];
  const float* lng  = (const float*)d_in[1];
  const float* lnb  = (const float*)d_in[2];
  const float* attw = (const float*)d_in[3];
  const float* attb = (const float*)d_in[4];
  const float* w1   = (const float*)d_in[5];
  const float* b1   = (const float*)d_in[6];
  const float* w2   = (const float*)d_in[7];
  const float* b2   = (const float*)d_in[8];
  const float* rww  = (const float*)d_in[9];
  const float* cww  = (const float*)d_in[10];
  float* out = (float*)d_out;
  uint8_t* ws = (uint8_t*)d_ws;

  const size_t NT = 65536, C = 256;
  const size_t MB = 1024ull*1024ull;
  float*  y   = (float*)(ws);                         // 64 MiB
  float*  xs  = (float*)(ws + 64*MB);                 // 64 MiB
  __bf16* ln  = (__bf16*)(ws + 128*MB);               // 32 MiB
  __bf16* qb  = (__bf16*)(ws + 160*MB);               // union region: 128 MiB
  __bf16* kb  = qb + NT*C;
  __bf16* vb  = kb + NT*C;
  __bf16* ao  = vb + NT*C;
  __bf16* hb  = qb;                                   // (65536x1024) aliases q/k/v/ao
  __bf16* wat = (__bf16*)(ws + 288*MB);               // transposed bf16 weights
  __bf16* wm1 = wat + 786432;                         // (2) 1024x256
  __bf16* wm2 = wm1 + 524288;                         // (2) 256x1024
  __bf16* Pb  = (__bf16*)(ws + 292*MB);               // self-attn P

  auto gemm = [&](const __bf16* A, const __bf16* Bm, const float* bias,
                  float* of, __bf16* ob, int M, int N, int K, int mode){
    k_gemm<<<dim3(M/128, N/64), 256, 0, stream>>>(A, Bm, bias, of, ob, M, N, K, mode);
  };

  // weights -> bf16, transposed to N-major
  for (int m = 0; m < 12; ++m)
    k_f2bT<<<dim3(8,8), dim3(32,8), 0, stream>>>(attw + m*65536, wat + m*65536, 256, 256);
  for (int m = 0; m < 2; ++m)
    k_f2bT<<<dim3(32,8), dim3(32,8), 0, stream>>>(w1 + m*262144, wm1 + m*262144, 256, 1024);
  for (int m = 0; m < 2; ++m)
    k_f2bT<<<dim3(8,32), dim3(32,8), 0, stream>>>(w2 + m*262144, wm2 + m*262144, 1024, 256);

  // x0 residual stream
  k_chw2tc<<<dim3(128,8,16), dim3(32,8), 0, stream>>>(x, y);

  // ----- self attention (from clean x0) -> xs -----
  k_ln<<<8192,256,0,stream>>>(y, lng + 0*C, lnb + 0*C, ln, 1e-5f, 0);
  gemm(ln, wat + (2*4+0)*65536, attb + (2*4+0)*256, nullptr, qb, 65536,256,256, 2);
  gemm(ln, wat + (2*4+1)*65536, attb + (2*4+1)*256, nullptr, kb, 65536,256,256, 2);
  gemm(ln, wat + (2*4+2)*65536, attb + (2*4+2)*256, nullptr, vb, 65536,256,256, 2);
  k_sattn_score<<<128,256,0,stream>>>(qb, kb, Pb);
  k_sattn_apply<<<65536,256,0,stream>>>(vb, Pb, ao);
  gemm(ao, wat + (2*4+3)*65536, attb + (2*4+3)*256, xs, nullptr, 65536,256,256, 0);

  // ----- row attention, accumulate into y -----
  k_ln<<<8192,256,0,stream>>>(y, lng + 2*C, lnb + 2*C, ln, 1e-4f, 0);
  gemm(ln, wat + (0*4+0)*65536, attb + (0*4+0)*256, nullptr, qb, 65536,256,256, 2);
  gemm(ln, wat + (0*4+1)*65536, attb + (0*4+1)*256, nullptr, kb, 65536,256,256, 2);
  gemm(ln, wat + (0*4+2)*65536, attb + (0*4+2)*256, nullptr, vb, 65536,256,256, 2);
  k_rcattn<<<8192,128,0,stream>>>(qb, kb, vb, rww, ao, 0);
  gemm(ao, wat + (0*4+3)*65536, attb + (0*4+3)*256, y, nullptr, 65536,256,256, 1);

  // ----- col attention (bug-faithful gather from x), accumulate into y -----
  k_ln<<<8192,256,0,stream>>>(x, lng + 3*C, lnb + 3*C, ln, 1e-4f, 1);
  gemm(ln, wat + (1*4+0)*65536, attb + (1*4+0)*256, nullptr, qb, 65536,256,256, 2);
  gemm(ln, wat + (1*4+1)*65536, attb + (1*4+1)*256, nullptr, kb, 65536,256,256, 2);
  gemm(ln, wat + (1*4+2)*65536, attb + (1*4+2)*256, nullptr, vb, 65536,256,256, 2);
  k_rcattn<<<8192,128,0,stream>>>(qb, kb, vb, cww, ao, 1);
  gemm(ao, wat + (1*4+3)*65536, attb + (1*4+3)*256, y, nullptr, 65536,256,256, 1);

  // ----- MLP1 (GELU2) -----
  k_ln<<<8192,256,0,stream>>>(y, lng + 4*C, lnb + 4*C, ln, 1e-4f, 0);
  gemm(ln, wm1 + 0,      b1 + 0,    nullptr, hb, 65536,1024,256, 3);
  gemm(hb, wm2 + 0,      b2 + 0,    y, nullptr,  65536,256,1024, 1);

  // y += xs
  k_add<<<65536,256,0,stream>>>(y, xs, 16777216);

  // ----- MLP2 (exact GELU) -----
  k_ln<<<8192,256,0,stream>>>(y, lng + 1*C, lnb + 1*C, ln, 1e-5f, 0);
  gemm(ln, wm1 + 262144, b1 + 1024, nullptr, hb, 65536,1024,256, 4);
  gemm(hb, wm2 + 262144, b2 + 1024, y, nullptr,  65536,256,1024, 1);

  // back to (B,C,H,W)
  k_tc2chw<<<dim3(128,8,16), dim3(32,8), 0, stream>>>(y, out);
}
